// MLP_60206851556132
// MI455X (gfx1250) — compile-verified, matrix operand order checked by reading, not run on hardware
//
#include <hip/hip_runtime.h>

// ---------------------------------------------------------------------------
// Fused 9-layer NeRF MLP for MI455X (gfx1250, wave32, WMMA).
//
// Roofline: ~252 GFLOP vs ~45 MB of HBM traffic -> compute bound; all GEMMs
// run on v_wmma_f32_16x16x32_f16. Weights are pre-packed (f32->f16) into the
// exact per-lane WMMA B-fragment layout so the inner loop is just
// global_load_b128 + v_wmma. Activations stay in LDS for the whole network.
//
// v2: each wave now owns M=32 rows (two 16-row strips). Every B fragment is
// loaded once and used by two WMMAs -> halves packed-weight L2 traffic
// (~16 GB -> ~8 GB) and gives two independent accumulation chains per n-tile
// to cover WMMA latency.
// ---------------------------------------------------------------------------

typedef __attribute__((ext_vector_type(16))) _Float16 v16h;
typedef __attribute__((ext_vector_type(8)))  _Float16 v8h;
typedef __attribute__((ext_vector_type(8)))  float    v8f;

namespace {
constexpr int kInDim        = 39;
constexpr int kOutDim       = 4;
constexpr int kActStride    = 264;  // halves: 256 + 8 pad (16B) -> conflict-free ds_load_b128
constexpr int kXbStride     = 64;   // halves: padded 39 -> 64
constexpr int kRowsPerWave  = 32;   // two 16-row strips per wave
constexpr int kWavesPerBlk  = 4;    // blockDim = 128
constexpr int kRowsPerBlock = 128;  // 4 waves x 32 rows
// packed-weight offsets in halves (fragment = 32x16 K x N = 512 halves)
constexpr size_t kLOff0 = 0;        // fc1: Kp=2,  Np=16
constexpr size_t kLOff1 = 16384;    // fc2: Kp=8,  Np=16
constexpr size_t kLOff2 = 81920;    // fc3
constexpr size_t kLOff3 = 147456;   // fc4
constexpr size_t kLOff4 = 212992;   // fc5: Kp=10, Np=16
constexpr size_t kLOff5 = 294912;   // fc6
constexpr size_t kLOff6 = 360448;   // fc7
constexpr size_t kLOff7 = 425984;   // fc8
constexpr size_t kLOff8 = 491520;   // fc9: Kp=8,  Np=1
constexpr size_t kWPackHalves = 495616;          // total packed weight halves
constexpr size_t kWChunks     = kWPackHalves/16; // 16-half chunks (one lane-slice)
}

__device__ __forceinline__ v16h cat16(v8h lo, v8h hi) {
  return __builtin_shufflevector(lo, hi, 0,1,2,3,4,5,6,7,8,9,10,11,12,13,14,15);
}

// A-fragment (16x32, f16) from LDS. ISA 7.12.2: lane L holds row M=L%16,
// halves j -> K = kTile*32 + j + 8*(j>=8) + 8*(L>=16). The two 8-half groups
// are K-contiguous -> two ds_load_b128 per lane.
__device__ __forceinline__ v16h load_afrag(const _Float16* strip, int lane,
                                           int kTile, int stride) {
  const int m    = lane & 15;
  const int koff = kTile * 32 + ((lane & 16) ? 8 : 0);
  const _Float16* p = strip + m * stride + koff;
  v8h lo = *(const v8h*)(p);
  v8h hi = *(const v8h*)(p + 16);
  return cat16(lo, hi);
}

// B-fragment: pre-swizzled by the pack kernel, 16 contiguous halves per lane.
__device__ __forceinline__ v16h load_bfrag(const _Float16* p) {
  v8h lo = *(const v8h*)(p);
  v8h hi = *(const v8h*)(p + 8);
  return cat16(lo, hi);
}

// Epilogue: C tile (f32) + bias -> ReLU -> f16 into LDS strip.
// C layout: lane L, vgpr r -> (M = r + 8*(L>=16), N = nt*16 + L%16).
__device__ __forceinline__ void store_ctile(v8f c, float bb, _Float16* strip,
                                            int nt, int nlo, int mb) {
#pragma unroll
  for (int r = 0; r < 8; ++r) {
    float v = c[r] + bb;
    v = v > 0.0f ? v : 0.0f;  // ReLU
    strip[(mb + r) * kActStride + nt * 16 + nlo] = (_Float16)v;
  }
}

// One dense layer for this wave's two 16-row strips: out = relu(A @ W + b),
// N = 256 (16 n-tiles). A-fragments for both strips already in registers;
// each B fragment is loaded once and feeds both strips (2x B reuse + two
// independent accumulation chains).
template <int KP>
__device__ __forceinline__ void layer_tiles2(const v16h* a0, const v16h* a1,
                                             const _Float16* __restrict__ wl,
                                             const float* __restrict__ bl,
                                             _Float16* s0, _Float16* s1,
                                             int lane) {
  const int nlo = lane & 15;
  const int mb  = (lane & 16) ? 8 : 0;
#pragma unroll 1
  for (int nt = 0; nt < 16; ++nt) {
    v8f c0 = {};
    v8f c1 = {};
    const _Float16* wp = wl + (size_t)nt * KP * 512 + lane * 16;
#pragma unroll
    for (int k = 0; k < KP; ++k) {
      v16h b = load_bfrag(wp + (size_t)k * 512);
      c0 = __builtin_amdgcn_wmma_f32_16x16x32_f16(false, a0[k], false, b,
                                                  (short)0, c0, false, false);
      c1 = __builtin_amdgcn_wmma_f32_16x16x32_f16(false, a1[k], false, b,
                                                  (short)0, c1, false, false);
    }
    const float bb = bl[nt * 16 + nlo];
    store_ctile(c0, bb, s0, nt, nlo, mb);
    store_ctile(c1, bb, s1, nt, nlo, mb);
  }
}

#define LOAD_A_PAIR_ACT()                                             \
  _Pragma("unroll") for (int k = 0; k < 8; ++k) {                     \
    a0[k] = load_afrag(s0, lane, k, kActStride);                      \
    a1[k] = load_afrag(s1, lane, k, kActStride);                      \
  }

__global__ __launch_bounds__(128) void mlp_fused_kernel(
    const float* __restrict__ x, const _Float16* __restrict__ wpack,
    const float* __restrict__ bpack, float* __restrict__ out) {
  extern __shared__ _Float16 smem[];
  _Float16* act = smem;                                  // 128 x 264 halves
  _Float16* xb  = smem + kRowsPerBlock * kActStride;     // 128 x 64 halves

  const int lane    = threadIdx.x & 31;
  const int wave    = threadIdx.x >> 5;
  const int rowBase = blockIdx.x * kRowsPerBlock + wave * kRowsPerWave;
  _Float16* s0  = act + (size_t)(wave * kRowsPerWave) * kActStride;
  _Float16* s1  = s0 + 16 * kActStride;
  _Float16* xs0 = xb + (size_t)(wave * kRowsPerWave) * kXbStride;
  _Float16* xs1 = xs0 + 16 * kXbStride;

  // Stage this wave's 32 input rows: f32 -> f16, zero-pad cols 39..63.
  const float* xg = x + (size_t)rowBase * kInDim;
  for (int i = lane; i < kRowsPerWave * kXbStride; i += 32) {
    const int r = i >> 6, cidx = i & 63;
    const float v = (cidx < kInDim) ? xg[r * kInDim + cidx] : 0.0f;
    xs0[r * kXbStride + cidx] = (_Float16)v;
  }
  __syncthreads();

  v16h a0[10], a1[10];

  // fc1: K = 39 (padded 64) from xbuf
  a0[0] = load_afrag(xs0, lane, 0, kXbStride);
  a0[1] = load_afrag(xs0, lane, 1, kXbStride);
  a1[0] = load_afrag(xs1, lane, 0, kXbStride);
  a1[1] = load_afrag(xs1, lane, 1, kXbStride);
  layer_tiles2<2>(a0, a1, wpack + kLOff0, bpack + 0 * 256, s0, s1, lane);

  // fc2..fc4: K = 256
  LOAD_A_PAIR_ACT();
  layer_tiles2<8>(a0, a1, wpack + kLOff1, bpack + 1 * 256, s0, s1, lane);
  LOAD_A_PAIR_ACT();
  layer_tiles2<8>(a0, a1, wpack + kLOff2, bpack + 2 * 256, s0, s1, lane);
  LOAD_A_PAIR_ACT();
  layer_tiles2<8>(a0, a1, wpack + kLOff3, bpack + 3 * 256, s0, s1, lane);

  // fc5: concat([h, x]) -> K = 295 (padded 320); k-tiles 8..9 come from xbuf
  LOAD_A_PAIR_ACT();
  a0[8] = load_afrag(xs0, lane, 0, kXbStride);
  a0[9] = load_afrag(xs0, lane, 1, kXbStride);
  a1[8] = load_afrag(xs1, lane, 0, kXbStride);
  a1[9] = load_afrag(xs1, lane, 1, kXbStride);
  layer_tiles2<10>(a0, a1, wpack + kLOff4, bpack + 4 * 256, s0, s1, lane);

  // fc6..fc8
  LOAD_A_PAIR_ACT();
  layer_tiles2<8>(a0, a1, wpack + kLOff5, bpack + 5 * 256, s0, s1, lane);
  LOAD_A_PAIR_ACT();
  layer_tiles2<8>(a0, a1, wpack + kLOff6, bpack + 6 * 256, s0, s1, lane);
  LOAD_A_PAIR_ACT();
  layer_tiles2<8>(a0, a1, wpack + kLOff7, bpack + 7 * 256, s0, s1, lane);

  // fc9: N = 4 (padded 16), no ReLU, f32 out straight to global
  LOAD_A_PAIR_ACT();
  v8f c0 = {};
  v8f c1 = {};
  const _Float16* wp = wpack + kLOff8 + lane * 16;
#pragma unroll
  for (int k = 0; k < 8; ++k) {
    v16h b = load_bfrag(wp + (size_t)k * 512);
    c0 = __builtin_amdgcn_wmma_f32_16x16x32_f16(false, a0[k], false, b,
                                                (short)0, c0, false, false);
    c1 = __builtin_amdgcn_wmma_f32_16x16x32_f16(false, a1[k], false, b,
                                                (short)0, c1, false, false);
  }
  const int nlo = lane & 15;
  const int mb  = (lane & 16) ? 8 : 0;
  if (nlo < kOutDim) {
    const float bb = bpack[8 * 256 + nlo];
#pragma unroll
    for (int r = 0; r < 8; ++r) {
      out[(size_t)(rowBase + mb + r) * kOutDim + nlo]      = c0[r] + bb;
      out[(size_t)(rowBase + 16 + mb + r) * kOutDim + nlo] = c1[r] + bb;
    }
  }
}

// ---------------------------------------------------------------------------
// Pack kernel: f32 row-major weights -> f16 WMMA B-fragment order in d_ws.
// Fragment (kT, nT) of layer l lives at lOff[l] + (nT*Kp + kT)*512 halves;
// lane L's 16 halves are contiguous: element j holds
//   W[kT*32 + j + 8*(j>=8) + 8*(L>=16)][nT*16 + L%16]   (zero if OOB pad).
// Biases padded to 256 f32 per layer, appended after the weights.
// ---------------------------------------------------------------------------
__global__ void pack_params_kernel(
    const float* w1, const float* b1, const float* w2, const float* b2,
    const float* w3, const float* b3, const float* w4, const float* b4,
    const float* w5, const float* b5, const float* w6, const float* b6,
    const float* w7, const float* b7, const float* w8, const float* b8,
    const float* w9, const float* b9,
    _Float16* __restrict__ wpack, float* __restrict__ bpack) {
  const float* W[9] = {w1, w2, w3, w4, w5, w6, w7, w8, w9};
  const float* B[9] = {b1, b2, b3, b4, b5, b6, b7, b8, b9};
  const int DIN[9]  = {39, 256, 256, 256, 295, 256, 256, 256, 256};
  const int DOUT[9] = {256, 256, 256, 256, 256, 256, 256, 256, 4};
  const int KP[9]   = {2, 8, 8, 8, 10, 8, 8, 8, 8};
  const size_t CB[10] = {0, 1024, 5120, 9216, 13312, 18432,
                         22528, 26624, 30720, 30976};  // kLOff/16

  const size_t g = (size_t)blockIdx.x * blockDim.x + threadIdx.x;
  if (g < kWChunks) {
    int l = 0;
    while (g >= CB[l + 1]) ++l;
    const size_t chunk = g - CB[l];
    const int    lane  = (int)(chunk & 31);
    const size_t frag  = chunk >> 5;
    const int    kT    = (int)(frag % (size_t)KP[l]);
    const int    nT    = (int)(frag / (size_t)KP[l]);
    const int    n     = nT * 16 + (lane & 15);
    const int    kbase = kT * 32 + ((lane & 16) ? 8 : 0);
    const float* Wl    = W[l];
    const int din = DIN[l], dout = DOUT[l];
    _Float16* dst = wpack + g * 16;
#pragma unroll
    for (int j = 0; j < 16; ++j) {
      const int k = kbase + j + ((j >= 8) ? 8 : 0);
      const float v = (k < din && n < dout) ? Wl[(size_t)k * dout + n] : 0.0f;
      dst[j] = (_Float16)v;
    }
  } else if (g < kWChunks + 9 * 256) {
    const int i = (int)(g - kWChunks);
    const int l = i >> 8, cdx = i & 255;
    bpack[i] = (cdx < DOUT[l]) ? B[l][cdx] : 0.0f;
  }
}

extern "C" void kernel_launch(void* const* d_in, const int* in_sizes, int n_in,
                              void* d_out, int out_size, void* d_ws, size_t ws_size,
                              hipStream_t stream) {
  const float* x = (const float*)d_in[0];
  const float* w[9];
  const float* b[9];
  for (int i = 0; i < 9; ++i) {
    w[i] = (const float*)d_in[1 + 2 * i];
    b[i] = (const float*)d_in[2 + 2 * i];
  }
  _Float16* wpack = (_Float16*)d_ws;
  float*    bpack = (float*)((char*)d_ws + kWPackHalves * sizeof(_Float16));

  // Pack weights/biases (deterministic, reads only d_in, writes d_ws).
  const int packThreads = (int)(kWChunks + 9 * 256);
  pack_params_kernel<<<(packThreads + 255) / 256, 256, 0, stream>>>(
      w[0], b[0], w[1], b[1], w[2], b[2], w[3], b[3], w[4], b[4],
      w[5], b[5], w[6], b[6], w[7], b[7], w[8], b[8], wpack, bpack);

  // Fused MLP: 262144 points / 128 per block = 2048 blocks of 128 threads.
  const size_t smemBytes =
      (size_t)(kRowsPerBlock * kActStride + kRowsPerBlock * kXbStride) *
      sizeof(_Float16);  // 83968 B (< 320 KB/WGP)
  (void)hipFuncSetAttribute((const void*)mlp_fused_kernel,
                            hipFuncAttributeMaxDynamicSharedMemorySize,
                            (int)smemBytes);
  mlp_fused_kernel<<<262144 / kRowsPerBlock, 128, smemBytes, stream>>>(
      x, wpack, bpack, (float*)d_out);
}
